// LlamaMoC_triton_6579889898127
// MI455X (gfx1250) — compile-verified, matrix operand order checked by reading, not run on hardware
//
#include <hip/hip_runtime.h>

#define HIDDEN 4096
#define INTER  11008
#define NTOK   4096   // B*S = 2*2048
#define KACT   2048

typedef __bf16 bf16;
typedef __attribute__((ext_vector_type(16))) __bf16 v16bf;
typedef __attribute__((ext_vector_type(8)))  __bf16 v8bf;
typedef __attribute__((ext_vector_type(8)))  float  v8f;
typedef __attribute__((ext_vector_type(4)))  float  v4f;
typedef __attribute__((ext_vector_type(4)))  int    v4i;

#define BM 128
#define BN 128
#define BK 32
#define LDSS 40   // padded LDS row stride (bf16 elems): 80B rows spread banks

// ---------------------------------------------------------------------------
// gfx1250 async copy (memory -> LDS, ASYNCcnt-tracked)
// ---------------------------------------------------------------------------
#if defined(__has_builtin)
#  if __has_builtin(__builtin_amdgcn_global_load_async_to_lds_b128)
#    define HAVE_ASYNC_B128 1
#  endif
#  if __has_builtin(__builtin_amdgcn_s_wait_asynccnt)
#    define HAVE_WAIT_ASYNC 1
#  endif
#endif

typedef __attribute__((address_space(1))) v4i gas_v4i;
typedef __attribute__((address_space(3))) v4i las_v4i;

__device__ __forceinline__ void async_b128(const bf16* g, bf16* l) {
#ifdef HAVE_ASYNC_B128
    __builtin_amdgcn_global_load_async_to_lds_b128(
        (gas_v4i*)(void*)(g), (las_v4i*)(void*)(l), 0, 0);
#else
    unsigned loff = (unsigned)(unsigned long long)l;   // low 32 bits = LDS offset
    asm volatile("global_load_async_to_lds_b128 %0, %1, off"
                 :: "v"(loff), "v"(g) : "memory");
#endif
}

template <int N>
__device__ __forceinline__ void wait_async() {
#ifdef HAVE_WAIT_ASYNC
    __builtin_amdgcn_s_wait_asynccnt(N);
#else
    asm volatile("s_wait_asynccnt %0" :: "i"(N) : "memory");
#endif
}

// ---------------------------------------------------------------------------
// WMMA helpers
// ---------------------------------------------------------------------------
__device__ __forceinline__ v8f wmma_bf16(v16bf a, v16bf b, v8f c) {
    return __builtin_amdgcn_wmma_f32_16x16x32_bf16(
        false, a, false, b, (short)0, c, false, false);
}

// ISA 7.12.2 fragment: lane<16 holds row lane, ks {0..7,16..23};
// lane>=16 holds row lane-16, ks {8..15,24..31}.
__device__ __forceinline__ v16bf load_frag(const bf16* __restrict__ lds, int row0) {
    const int lane = threadIdx.x & 31;
    const bf16* base = lds + (row0 + (lane & 15)) * LDSS + ((lane >> 4) << 3);
    v8bf lo = *(const v8bf*)(base);
    v8bf hi = *(const v8bf*)(base + 16);
    return __builtin_shufflevector(lo, hi, 0, 1, 2, 3, 4, 5, 6, 7,
                                           8, 9, 10, 11, 12, 13, 14, 15);
}

// ---------------------------------------------------------------------------
// Kernel 0: streaming f32 -> bf16 (x and the three weight matrices)
// ---------------------------------------------------------------------------
__global__ __launch_bounds__(256) void cvt_f32_bf16(const float* __restrict__ x,
                                                    bf16* __restrict__ y) {
    const size_t i = ((size_t)blockIdx.x * 256 + threadIdx.x) * 8;
    v4f a = __builtin_nontemporal_load((const v4f*)(x + i));
    v4f b = __builtin_nontemporal_load((const v4f*)(x + i + 4));
    v8bf r;
    r[0] = (__bf16)a[0]; r[1] = (__bf16)a[1]; r[2] = (__bf16)a[2]; r[3] = (__bf16)a[3];
    r[4] = (__bf16)b[0]; r[5] = (__bf16)b[1]; r[6] = (__bf16)b[2]; r[7] = (__bf16)b[3];
    __builtin_nontemporal_store(r, (v8bf*)(y + i));
}

// ---------------------------------------------------------------------------
// Kernel 1: fused gate/up GEMM + SiLU, async double-buffered staging.
//   gate_out[t,n] = x @ gate_w^T   (f32, for exact top-k)
//   p_out[t,n]    = silu(gate) * (x @ up_w^T)   (bf16)
// ---------------------------------------------------------------------------
__global__ __launch_bounds__(256) void moc_gate_up(const bf16* __restrict__ xb,
                                                   const bf16* __restrict__ gwb,
                                                   const bf16* __restrict__ uwb,
                                                   float* __restrict__ gate_out,
                                                   bf16*  __restrict__ p_out) {
    __shared__ __align__(16) bf16 lA [2][BM * LDSS];
    __shared__ __align__(16) bf16 lBg[2][BN * LDSS];
    __shared__ __align__(16) bf16 lBu[2][BN * LDSS];

    const int tid  = threadIdx.x;
    const int lane = tid & 31;
    const int wid  = tid >> 5;
    const int wm   = wid & 1;          // 2 waves over M (64 rows)
    const int wn   = wid >> 1;         // 4 waves over N (32 cols)
    const int mbase = blockIdx.x * BM;
    const int nbase = blockIdx.y * BN;

    // staging: thread copies 16B chunks c0=tid, c1=tid+256 (row=c>>2, q=(c&3)*8)
    const int r0 = tid >> 2;
    const int r1 = r0 + 64;
    const int q  = (tid & 3) << 3;
    const int l0 = r0 * LDSS + q;
    const int l1 = r1 * LDSS + q;
    const bf16* a0 = xb  + (size_t)(mbase + r0) * HIDDEN + q;
    const bf16* a1 = xb  + (size_t)(mbase + r1) * HIDDEN + q;
    const bf16* g0 = gwb + (size_t)(nbase + r0) * HIDDEN + q;
    const bf16* g1 = gwb + (size_t)(nbase + r1) * HIDDEN + q;
    const bf16* u0 = uwb + (size_t)(nbase + r0) * HIDDEN + q;
    const bf16* u1 = uwb + (size_t)(nbase + r1) * HIDDEN + q;

    auto issue = [&](int buf, int kofs) {
        async_b128(a0 + kofs, &lA [buf][l0]);
        async_b128(a1 + kofs, &lA [buf][l1]);
        async_b128(g0 + kofs, &lBg[buf][l0]);
        async_b128(g1 + kofs, &lBg[buf][l1]);
        async_b128(u0 + kofs, &lBu[buf][l0]);
        async_b128(u1 + kofs, &lBu[buf][l1]);
    };

    v8f accG[4][2], accV[4][2];
    const v8f vzero = {0.f, 0.f, 0.f, 0.f, 0.f, 0.f, 0.f, 0.f};
    for (int i = 0; i < 4; ++i)
        for (int j = 0; j < 2; ++j) { accG[i][j] = vzero; accV[i][j] = vzero; }

    const int KSTEPS = HIDDEN / BK;            // 128
    issue(0, 0);
    for (int kb = 0; kb < KSTEPS; ++kb) {
        const int cur = kb & 1;
        if (kb + 1 < KSTEPS) { issue(cur ^ 1, (kb + 1) * BK); wait_async<6>(); }
        else                 { wait_async<0>(); }
        __syncthreads();                       // stage kb visible from all waves

        v16bf a[4], bg[2], bu[2];
#pragma unroll
        for (int i = 0; i < 4; ++i) a[i]  = load_frag(lA[cur],  wm * 64 + i * 16);
#pragma unroll
        for (int j = 0; j < 2; ++j) { bg[j] = load_frag(lBg[cur], wn * 32 + j * 16);
                                      bu[j] = load_frag(lBu[cur], wn * 32 + j * 16); }
#pragma unroll
        for (int i = 0; i < 4; ++i)
#pragma unroll
            for (int j = 0; j < 2; ++j) {
                accG[i][j] = wmma_bf16(a[i], bg[j], accG[i][j]);
                accV[i][j] = wmma_bf16(a[i], bu[j], accV[i][j]);
            }
        __syncthreads();                       // done reading buf before overwrite
    }

    // epilogue: D elem e -> row = e + 8*(lane>>4), col = lane&15
    const int rlane = (lane >> 4) << 3;
    const int clane = lane & 15;
#pragma unroll
    for (int i = 0; i < 4; ++i)
#pragma unroll
        for (int j = 0; j < 2; ++j) {
            const int col = nbase + wn * 32 + j * 16 + clane;
#pragma unroll
            for (int e = 0; e < 8; ++e) {
                const int row = mbase + wm * 64 + i * 16 + rlane + e;
                const float g  = accG[i][j][e];
                const float vv = accV[i][j][e];
                const float s  = g / (1.0f + __expf(-g));   // silu(g)
                gate_out[(size_t)row * INTER + col] = g;
                p_out  [(size_t)row * INTER + col] = (__bf16)(s * vv);
            }
        }
}

// ---------------------------------------------------------------------------
// Kernel 2: exact per-token radix select (2048th-largest gate), mask p.
// ---------------------------------------------------------------------------
__global__ __launch_bounds__(256) void topk_mask(const float* __restrict__ gate,
                                                 bf16* __restrict__ p) {
    __shared__ unsigned keys[INTER];           // 44 KB of the 320 KB WGP LDS
    __shared__ unsigned hist[256];
    __shared__ unsigned s_bin, s_above;

    const int tid = threadIdx.x;
    const size_t base = (size_t)blockIdx.x * INTER;

    for (int i = tid; i < INTER; i += 256) {   // 43 uniform iterations
        unsigned b = __float_as_uint(gate[base + i]);
        keys[i] = (b & 0x80000000u) ? ~b : (b | 0x80000000u);
    }

    unsigned prefix = 0;
    int k = KACT;
    for (int shift = 24; shift >= 0; shift -= 8) {
        hist[tid] = 0;
        __syncthreads();
        const unsigned pmask = (shift == 24) ? 0u : (0xFFFFFFFFu << (shift + 8));
        for (int i = tid; i < INTER; i += 256) {
            const unsigned key = keys[i];
            if ((key & pmask) == prefix)
                atomicAdd(&hist[(key >> shift) & 255u], 1u);
        }
        __syncthreads();
        if (tid == 0) {
            unsigned cum = 0, above = 0, bin = 0;
            for (int b2 = 255; b2 >= 0; --b2) {
                const unsigned h = hist[b2];
                if (cum + h >= (unsigned)k) { bin = (unsigned)b2; above = cum; break; }
                cum += h;
            }
            s_bin = bin; s_above = above;
        }
        __syncthreads();
        prefix |= (s_bin << shift);
        k -= (int)s_above;
        __syncthreads();
    }

    const unsigned kth = prefix;
    for (int i = tid; i < INTER; i += 256)
        if (keys[i] < kth) p[base + i] = (__bf16)0.0f;
}

// ---------------------------------------------------------------------------
// Kernel 3: down GEMM.  out[t,h] = sum_e p[t,e] * down_w[h,e]
// ---------------------------------------------------------------------------
__global__ __launch_bounds__(256) void moc_down(const bf16* __restrict__ pmat,
                                                const bf16* __restrict__ dwb,
                                                float* __restrict__ out) {
    __shared__ __align__(16) bf16 lA[2][BM * LDSS];
    __shared__ __align__(16) bf16 lB[2][BN * LDSS];

    const int tid  = threadIdx.x;
    const int lane = tid & 31;
    const int wid  = tid >> 5;
    const int wm   = wid & 1;
    const int wn   = wid >> 1;
    const int mbase = blockIdx.x * BM;
    const int nbase = blockIdx.y * BN;

    const int r0 = tid >> 2;
    const int r1 = r0 + 64;
    const int q  = (tid & 3) << 3;
    const int l0 = r0 * LDSS + q;
    const int l1 = r1 * LDSS + q;
    const bf16* a0 = pmat + (size_t)(mbase + r0) * INTER + q;
    const bf16* a1 = pmat + (size_t)(mbase + r1) * INTER + q;
    const bf16* b0 = dwb  + (size_t)(nbase + r0) * INTER + q;
    const bf16* b1 = dwb  + (size_t)(nbase + r1) * INTER + q;

    auto issue = [&](int buf, int kofs) {
        async_b128(a0 + kofs, &lA[buf][l0]);
        async_b128(a1 + kofs, &lA[buf][l1]);
        async_b128(b0 + kofs, &lB[buf][l0]);
        async_b128(b1 + kofs, &lB[buf][l1]);
    };

    v8f acc[4][2];
    const v8f vzero = {0.f, 0.f, 0.f, 0.f, 0.f, 0.f, 0.f, 0.f};
    for (int i = 0; i < 4; ++i)
        for (int j = 0; j < 2; ++j) acc[i][j] = vzero;

    const int KSTEPS = INTER / BK;             // 344
    issue(0, 0);
    for (int kb = 0; kb < KSTEPS; ++kb) {
        const int cur = kb & 1;
        if (kb + 1 < KSTEPS) { issue(cur ^ 1, (kb + 1) * BK); wait_async<4>(); }
        else                 { wait_async<0>(); }
        __syncthreads();

        v16bf a[4], b[2];
#pragma unroll
        for (int i = 0; i < 4; ++i) a[i] = load_frag(lA[cur], wm * 64 + i * 16);
#pragma unroll
        for (int j = 0; j < 2; ++j) b[j] = load_frag(lB[cur], wn * 32 + j * 16);
#pragma unroll
        for (int i = 0; i < 4; ++i)
#pragma unroll
            for (int j = 0; j < 2; ++j)
                acc[i][j] = wmma_bf16(a[i], b[j], acc[i][j]);
        __syncthreads();
    }

    const int rlane = (lane >> 4) << 3;
    const int clane = lane & 15;
#pragma unroll
    for (int i = 0; i < 4; ++i)
#pragma unroll
        for (int j = 0; j < 2; ++j) {
            const int col = nbase + wn * 32 + j * 16 + clane;
#pragma unroll
            for (int e = 0; e < 8; ++e) {
                const int row = mbase + wm * 64 + i * 16 + rlane + e;
                out[(size_t)row * HIDDEN + col] = acc[i][j][e];
            }
        }
}

// ---------------------------------------------------------------------------
// host launcher
// ---------------------------------------------------------------------------
extern "C" void kernel_launch(void* const* d_in, const int* in_sizes, int n_in,
                              void* d_out, int out_size, void* d_ws, size_t ws_size,
                              hipStream_t stream) {
    (void)in_sizes; (void)n_in; (void)out_size; (void)ws_size;
    const float* x      = (const float*)d_in[0];   // [2,2048,4096]
    const float* gate_w = (const float*)d_in[1];   // [11008,4096]
    const float* up_w   = (const float*)d_in[2];   // [11008,4096]
    const float* down_w = (const float*)d_in[3];   // [4096,11008]
    float*       out    = (float*)d_out;           // [2,2048,4096]

    char* ws = (char*)d_ws;
    const size_t NX = (size_t)NTOK * HIDDEN;       // 16,777,216
    const size_t NW = (size_t)INTER * HIDDEN;      // 45,088,768 (all 3 weights)
    bf16*  xb   = (bf16*)ws;                        ws += NX * sizeof(bf16);
    bf16*  gwb  = (bf16*)ws;                        ws += NW * sizeof(bf16);
    bf16*  uwb  = (bf16*)ws;                        ws += NW * sizeof(bf16);
    bf16*  dwb  = (bf16*)ws;                        ws += NW * sizeof(bf16);
    float* gate = (float*)ws;                       ws += (size_t)NTOK * INTER * sizeof(float);
    bf16*  p    = (bf16*)ws;

    // 0) one-shot f32 -> bf16 conversions (streaming, non-temporal)
    cvt_f32_bf16<<<NX / 2048, 256, 0, stream>>>(x, xb);
    cvt_f32_bf16<<<NW / 2048, 256, 0, stream>>>(gate_w, gwb);
    cvt_f32_bf16<<<NW / 2048, 256, 0, stream>>>(up_w, uwb);
    cvt_f32_bf16<<<NW / 2048, 256, 0, stream>>>(down_w, dwb);

    // 1) fused gate/up GEMM + silu (M fastest => weight panels reused in L2)
    dim3 g1(NTOK / BM, INTER / BN);                // 32 x 86
    moc_gate_up<<<g1, 256, 0, stream>>>(xb, gwb, uwb, gate, p);

    // 2) per-token exact top-2048 threshold + mask
    topk_mask<<<NTOK, 256, 0, stream>>>(gate, p);

    // 3) down projection
    dim3 g3(NTOK / BM, HIDDEN / BN);               // 32 x 32
    moc_down<<<g3, 256, 0, stream>>>(p, dwb, out);
}
